// MultiHeadedAttention_1614907703574
// MI455X (gfx1250) — matrix operand order repaired and checked by reading, not compiled, not run on hardware
//
#include <hip/hip_runtime.h>
#include <hip/hip_bf16.h>
#include <stdint.h>

#define B_  2
#define S_  2048
#define DM_ 1024
#define H_  16
#define D_  64

typedef __attribute__((ext_vector_type(16))) __bf16 bf16x16;
typedef __attribute__((ext_vector_type(8)))  float  v8f;

struct alignas(16) U128 { unsigned int x, y, z, w; };

union Frag {
  bf16x16 v;
  U128 q[2];
  unsigned short u[16];
};

__device__ __forceinline__ unsigned short f2bf(float f) {
  unsigned int u = __float_as_uint(f);
  u += 0x7FFFu + ((u >> 16) & 1u);   // round-to-nearest-even
  return (unsigned short)(u >> 16);
}

__device__ __forceinline__ v8f vzero8() {
  v8f z;
#pragma unroll
  for (int i = 0; i < 8; ++i) z[i] = 0.0f;
  return z;
}

__device__ __forceinline__ v8f wmma_bf16(bf16x16 a, bf16x16 b, v8f c) {
  return __builtin_amdgcn_wmma_f32_16x16x32_bf16(false, a, false, b, (short)0, c,
                                                 false, false);
}

// max across the 16 lanes of a DPP row, pure VALU (no LDS bpermute):
// xor1 -> xor2 -> half_mirror (4-groups) -> mirror (8-groups)
__device__ __forceinline__ float dpp_max16(float x) {
  int v = __float_as_int(x), t;
  t = __builtin_amdgcn_update_dpp(0, v, 0xB1,  0xF, 0xF, true); // quad_perm [1,0,3,2]
  v = __float_as_int(fmaxf(__int_as_float(v), __int_as_float(t)));
  t = __builtin_amdgcn_update_dpp(0, v, 0x4E,  0xF, 0xF, true); // quad_perm [2,3,0,1]
  v = __float_as_int(fmaxf(__int_as_float(v), __int_as_float(t)));
  t = __builtin_amdgcn_update_dpp(0, v, 0x141, 0xF, 0xF, true); // row_half_mirror
  v = __float_as_int(fmaxf(__int_as_float(v), __int_as_float(t)));
  t = __builtin_amdgcn_update_dpp(0, v, 0x140, 0xF, 0xF, true); // row_mirror
  v = __float_as_int(fmaxf(__int_as_float(v), __int_as_float(t)));
  return __int_as_float(v);
}

// A fragment (16x32 bf16) from row-major [rows x ld] bf16 matrix.
template <typename PT>
__device__ __forceinline__ bf16x16 load_a_bf16(const PT* base, int row0, int ld,
                                               int k0, int lane) {
  int m = lane & 15, half = lane >> 4;
  const PT* p = base + (row0 + m) * ld + k0;
  Frag f;
  f.q[0] = *(const U128*)(p + 16 * half);
  f.q[1] = *(const U128*)(p + 16 * half + 8);
  return f.v;
}

// B fragment (32x16 bf16) given Bt = B^T stored row-major [N x K].
__device__ __forceinline__ bf16x16 load_b_bt(const unsigned short* bt, int n0,
                                             int ld, int k0, int lane) {
  int n = lane & 15, half = lane >> 4;
  const unsigned short* p = bt + (n0 + n) * ld + k0 + 16 * half;
  Frag f;
  f.q[0] = *(const U128*)(p);
  f.q[1] = *(const U128*)(p + 8);
  return f.v;
}

// ---------------- Kernel 0: q/k/v fp32 -> bf16 (memory-bound, one pass) ----
__global__ __launch_bounds__(256) void cvt_kernel(
    const float* __restrict__ q, const float* __restrict__ k,
    const float* __restrict__ v, unsigned short* __restrict__ Xb) {
  int i = blockIdx.x * 256 + threadIdx.x;        // quad index < 4096*1024/4
  int w = blockIdx.y;
  const float* X = (w == 0) ? q : (w == 1) ? k : v;
  float4 fv = ((const float4*)X)[i];
  unsigned int lo = (unsigned)f2bf(fv.x) | ((unsigned)f2bf(fv.y) << 16);
  unsigned int hi = (unsigned)f2bf(fv.z) | ((unsigned)f2bf(fv.w) << 16);
  uint2 o; o.x = lo; o.y = hi;
  *(uint2*)(Xb + ((size_t)w << 22) + ((size_t)i << 2)) = o;
}

// ---------------- Kernel 1: weight transpose + convert to bf16 -------------
__global__ __launch_bounds__(256) void wtrans_kernel(
    const float* __restrict__ Wq, const float* __restrict__ Wk,
    const float* __restrict__ Wv, unsigned short* __restrict__ Wt) {
  int idx = blockIdx.x * 256 + threadIdx.x;      // n*1024 + k
  int w = blockIdx.y;
  const float* W = (w == 0) ? Wq : (w == 1) ? Wk : Wv;
  int nn = idx >> 10, kk = idx & 1023;
  Wt[(size_t)w * (DM_ * DM_) + idx] = f2bf(W[kk * DM_ + nn]);
}

// ---------------- Kernel 2: projection GEMM (WMMA, double-buffered) --------
// scale is folded into the stored bf16 result (Q gets 0.125*log2(e) so the
// attention scores land directly in the exp2 domain).
template <int TR>
__global__ __launch_bounds__(128) void proj_kernel(
    const unsigned short* __restrict__ X, const unsigned short* __restrict__ Wt,
    const float* __restrict__ bias, unsigned short* __restrict__ out,
    float scale) {
  int lane = threadIdx.x & 31, wave = threadIdx.x >> 5;
  int tile = blockIdx.x * 4 + wave;              // 4096 wave-tiles (16x64 each)
  int tn = tile & 15, tm = tile >> 4;
  int row0 = tm * 16, col0 = tn * 64;
  v8f c0 = vzero8(), c1 = vzero8(), c2 = vzero8(), c3 = vzero8();

  bf16x16 a  = load_a_bf16(X, row0, DM_, 0, lane);
  bf16x16 b0 = load_b_bt(Wt, col0,      DM_, 0, lane);
  bf16x16 b1 = load_b_bt(Wt, col0 + 16, DM_, 0, lane);
  bf16x16 b2 = load_b_bt(Wt, col0 + 32, DM_, 0, lane);
  bf16x16 b3 = load_b_bt(Wt, col0 + 48, DM_, 0, lane);

  for (int k0 = 0; k0 < DM_; k0 += 32) {
    int k1 = (k0 + 32) & (DM_ - 1);              // wraps on last iter (harmless)
    bf16x16 an  = load_a_bf16(X, row0, DM_, k1, lane);
    bf16x16 bn0 = load_b_bt(Wt, col0,      DM_, k1, lane);
    bf16x16 bn1 = load_b_bt(Wt, col0 + 16, DM_, k1, lane);
    bf16x16 bn2 = load_b_bt(Wt, col0 + 32, DM_, k1, lane);
    bf16x16 bn3 = load_b_bt(Wt, col0 + 48, DM_, k1, lane);
    c0 = wmma_bf16(a, b0, c0);
    c1 = wmma_bf16(a, b1, c1);
    c2 = wmma_bf16(a, b2, c2);
    c3 = wmma_bf16(a, b3, c3);
    a = an; b0 = bn0; b1 = bn1; b2 = bn2; b3 = bn3;
  }

  int n = lane & 15, mbase = (lane >> 4) * 8;    // C layout: col n, rows mbase+r
#pragma unroll
  for (int j = 0; j < 4; ++j) {
    v8f c = (j == 0) ? c0 : (j == 1) ? c1 : (j == 2) ? c2 : c3;
#pragma unroll
    for (int r = 0; r < 8; ++r) {
      int row = row0 + mbase + r;                // b*2048 + s
      int col = col0 + 16 * j + n;               // 0..1023
      unsigned short bv = f2bf((c[r] + bias[col]) * scale);
      if (TR == 0) {
        out[(size_t)row * (H_ * D_) + col] = bv;
      } else {
        int bb = row >> 11, s = row & 2047;
        int hh = s >> 7;                          // head = s/128 (faithful reshape)
        int t  = ((s & 127) << 4) + (col >> 6);   // position within head
        int d  = col & 63;
        out[(((size_t)(bb * H_ + hh) * D_ + d) << 11) + t] = bv;
      }
    }
  }
}

// ---------------- Kernel 3: flash attention per (b,h) ----------------------
// Scores arrive pre-scaled by 0.125*log2(e): softmax done in exp2 domain.
__global__ __launch_bounds__(128) void attn_kernel(
    const unsigned short* __restrict__ Qp, const unsigned short* __restrict__ Kp,
    const unsigned short* __restrict__ Vt, float* __restrict__ Oh) {
  __shared__ unsigned short ldsP[4 * 16 * 32];   // per-wave 16x32 bf16 P tile
  int lane = threadIdx.x & 31, wave = threadIdx.x >> 5;
  int b = blockIdx.z, h = blockIdx.y;
  int rb = blockIdx.x * 64 + wave * 16;          // 16 query rows per wave
  const unsigned short* Qh = Qp + ((size_t)b << 21) + ((size_t)h << 17);
  const unsigned short* Kh = Kp + ((size_t)b << 21) + ((size_t)h << 17);
  const unsigned short* Vh = Vt + ((size_t)(b * H_ + h) << 17);
  unsigned short* pbuf = ldsP + wave * 512;
  int n = lane & 15, mbase = (lane >> 4) * 8;

  bf16x16 qa0 = load_a_bf16(Qh, rb, D_, 0, lane);
  bf16x16 qa1 = load_a_bf16(Qh, rb, D_, 32, lane);
  v8f o0 = vzero8(), o1 = vzero8(), o2 = vzero8(), o3 = vzero8();
  v8f lsum = vzero8();                           // row sums via ones-WMMA
  Frag onesf;
#pragma unroll
  for (int i = 0; i < 16; ++i) onesf.u[i] = 0x3F80;  // bf16 1.0
  float M[8];
#pragma unroll
  for (int r = 0; r < 8; ++r) M[r] = -3.0e38f;

  // prime K-tile fragments for t0 = 0
  bf16x16 kb00 = load_b_bt(Kh, 0,  D_, 0,  lane);
  bf16x16 kb01 = load_b_bt(Kh, 0,  D_, 32, lane);
  bf16x16 kb10 = load_b_bt(Kh, 16, D_, 0,  lane);
  bf16x16 kb11 = load_b_bt(Kh, 16, D_, 32, lane);

  for (int t0 = 0; t0 < S_; t0 += 32) {
    __builtin_prefetch(Kh + ((t0 + 64) & (S_ - 1)) * D_, 0, 1);
    // S tile [16 x 32] = Q * K^T   (already in log2 domain)
    v8f s0 = wmma_bf16(qa0, kb00, vzero8());
    s0     = wmma_bf16(qa1, kb01, s0);
    v8f s1 = wmma_bf16(qa0, kb10, vzero8());
    s1     = wmma_bf16(qa1, kb11, s1);

    int t1 = (t0 + 32) & (S_ - 1);
    // preload next K tile + current V tile: overlaps with softmax VALU below
    kb00 = load_b_bt(Kh, t1,      D_, 0,  lane);
    kb01 = load_b_bt(Kh, t1,      D_, 32, lane);
    kb10 = load_b_bt(Kh, t1 + 16, D_, 0,  lane);
    kb11 = load_b_bt(Kh, t1 + 16, D_, 32, lane);
    bf16x16 vb0 = load_b_bt(Vh, 0,  S_, t0, lane);
    bf16x16 vb1 = load_b_bt(Vh, 16, S_, t0, lane);
    bf16x16 vb2 = load_b_bt(Vh, 32, S_, t0, lane);
    bf16x16 vb3 = load_b_bt(Vh, 48, S_, t0, lane);

    // online softmax over the 32 columns (exp2 domain, DPP row-max)
#pragma unroll
    for (int r = 0; r < 8; ++r) {
      float a = s0[r];
      float c = s1[r];
      float mx = dpp_max16(fmaxf(a, c));
      float newM = fmaxf(M[r], mx);
      float corr = __builtin_amdgcn_exp2f(M[r] - newM);
      float p0 = __builtin_amdgcn_exp2f(a - newM);
      float p1 = __builtin_amdgcn_exp2f(c - newM);
      M[r] = newM;
      lsum[r] *= corr;
      o0[r] *= corr; o1[r] *= corr; o2[r] *= corr; o3[r] *= corr;
      pbuf[(mbase + r) * 32 + n]      = f2bf(p0);
      pbuf[(mbase + r) * 32 + 16 + n] = f2bf(p1);
    }
    __builtin_amdgcn_wave_barrier();
    asm volatile("s_wait_dscnt 0" ::: "memory");
    // O tile [16 x 64] += P[16 x 32] * V[32 x 64] ; rowsum += P * 1
    bf16x16 pa = load_a_bf16(pbuf, 0, 32, 0, lane);
    o0 = wmma_bf16(pa, vb0, o0);
    o1 = wmma_bf16(pa, vb1, o1);
    o2 = wmma_bf16(pa, vb2, o2);
    o3 = wmma_bf16(pa, vb3, o3);
    lsum = wmma_bf16(pa, onesf.v, lsum);
    __builtin_amdgcn_wave_barrier();
  }

  float* obase = Oh + (size_t)((b * H_ + h) * S_ + rb) * D_;
#pragma unroll
  for (int r = 0; r < 8; ++r) {
    float inv = 1.0f / lsum[r];
    float* orow = obase + (mbase + r) * D_;
    orow[n]      = o0[r] * inv;
    orow[16 + n] = o1[r] * inv;
    orow[32 + n] = o2[r] * inv;
    orow[48 + n] = o3[r] * inv;
  }
}

// ---------------- Kernel 4: head-combine (deterministic, no atomics) -------
__global__ __launch_bounds__(256) void reduce_kernel(
    const float* __restrict__ Oh, const float* __restrict__ W0w,
    const float* __restrict__ W0b, float* __restrict__ out) {
  int i = blockIdx.x * 256 + threadIdx.x;        // < B*S*D = 262144
  int b = i >> 17;
  int sd = i & 131071;
  float acc = W0b[0];
  const float* p = Oh + (size_t)b * H_ * (S_ * D_) + sd;
#pragma unroll
  for (int h = 0; h < H_; ++h) acc += W0w[h] * p[(size_t)h * (S_ * D_)];
  out[i] = acc;
}

extern "C" void kernel_launch(void* const* d_in, const int* in_sizes, int n_in,
                              void* d_out, int out_size, void* d_ws, size_t ws_size,
                              hipStream_t stream) {
  const float* q    = (const float*)d_in[0];
  const float* k    = (const float*)d_in[1];
  const float* v    = (const float*)d_in[2];
  const float* Wq_w = (const float*)d_in[3];
  const float* Wq_b = (const float*)d_in[4];
  const float* Wk_w = (const float*)d_in[5];
  const float* Wk_b = (const float*)d_in[6];
  const float* Wv_w = (const float*)d_in[7];
  const float* Wv_b = (const float*)d_in[8];
  const float* W0_w = (const float*)d_in[9];
  const float* W0_b = (const float*)d_in[10];
  float* out = (float*)d_out;

  char* ws = (char*)d_ws;
  unsigned short* Wt = (unsigned short*)(ws);                       //  6 MB
  unsigned short* Xb = (unsigned short*)(ws + 6291456);             // 24 MB (qb,kb,vb)
  unsigned short* Qp = (unsigned short*)(ws + 31457280);            //  8 MB
  unsigned short* Kp = (unsigned short*)(ws + 39845888);            //  8 MB
  unsigned short* Vt = (unsigned short*)(ws + 48234496);            //  8 MB
  float*          Oh = (float*)(ws + 56623104);                     // 16 MB

  dim3 cv_grid(4096, 3);
  cvt_kernel<<<cv_grid, 256, 0, stream>>>(q, k, v, Xb);
  wtrans_kernel<<<cv_grid, 256, 0, stream>>>(Wq_w, Wk_w, Wv_w, Wt);

  // Q carries 1/sqrt(64) * log2(e) so attention scores are in the exp2 domain
  const float qscale = 0.1803368801111204f;
  proj_kernel<0><<<1024, 128, 0, stream>>>(Xb,           Wt,           Wq_b, Qp, qscale);
  proj_kernel<0><<<1024, 128, 0, stream>>>(Xb + 4194304, Wt + 1048576, Wk_b, Kp, 1.0f);
  proj_kernel<1><<<1024, 128, 0, stream>>>(Xb + 8388608, Wt + 2097152, Wv_b, Vt, 1.0f);

  dim3 at_grid(32, 16, 2);
  attn_kernel<<<at_grid, 128, 0, stream>>>(Qp, Kp, Vt, Oh);

  reduce_kernel<<<1024, 256, 0, stream>>>(Oh, W0_w, W0_b, out);
}
